// UniGRefDeformableHead_38259568673489
// MI455X (gfx1250) — matrix operand order repaired and checked by reading, not compiled, not run on hardware
//
#include <hip/hip_runtime.h>
#include <math.h>

// ---------------------------------------------------------------------------
// MI455X implementation of the DETR-style decoder head.
//  - cross-attn logits:  Qk(80x256) @ X^T  via v_wmma_f32_16x16x32_bf16
//  - attn @ mem:         attn(80x128chunk) @ memT via v_wmma_f32_16x16x32_bf16
//  - wk / wv are folded into Q / post-sum respectively => no K/V projections.
//  - two streaming passes over img_feat (~1.1 GB) => memory-bound, ~50us roofline.
//  - A/B fragments fully staged in LDS and preloaded into registers so the
//    8-deep WMMA accumulator chains issue without per-step memory waits.
// ---------------------------------------------------------------------------

#define HC   256
#define NQ   10
#define NHD  8
#define FFD  512
#define NB   32
#define NL   40
#define HWN  16384
#define IW   128
#define R80  80
#define NCH  128   // chunks of the HW dimension
#define CK   128   // positions per chunk

#define O1 0                  // pos_feat_out  (32*10*256)
#define O2 81920              // pos_points    (32*10*2)
#define O3 82560              // global_attn   (32*16384)
#define O4 606848             // attn          (32*10*16384)

#define LNEPS  1e-5f
#define SCALE  0.17677669529663687f   // 1/sqrt(32)

// per-batch packed A-fragment block: 5 r-tiles * 8 kc * 32 lanes * 16 bf16
#define QKFRAG_B 20480        // u16 elements per batch

typedef __attribute__((ext_vector_type(16))) __bf16        bf16x16;
typedef __attribute__((ext_vector_type(8)))  float         f32x8;
typedef __attribute__((ext_vector_type(4)))  unsigned int  u32x4;

union CFrag { f32x8 v; float f[8]; };
union BFrag { u32x4 u[2]; bf16x16 v; };

__device__ __forceinline__ unsigned short f2bf(float x) {
  unsigned u = __builtin_bit_cast(unsigned, x);
  u += 0x7FFFu + ((u >> 16) & 1u);          // round-to-nearest-even
  return (unsigned short)(u >> 16);
}

// A / B^T fragment from LDS: layout [row][K], row = R0 + lane%16,
// elem e -> K = K0 + (e/8)*16 + (lane/16)*8 + e%8   (16-bit WMMA operand layout)
__device__ __forceinline__ bf16x16 ldsFrag(const unsigned short* p, int S, int R0, int K0, int lane) {
  int row = R0 + (lane & 15), half = lane >> 4;
  BFrag f;
  const unsigned short* q = p + row * S + K0 + half * 8;
  f.u[0] = *(const u32x4*)q;
  f.u[1] = *(const u32x4*)(q + 16);
  return f.v;
}

// pre-packed fragment (16 contiguous bf16 per lane) from LDS
__device__ __forceinline__ bf16x16 ldsFragC(const unsigned short* base, int lane) {
  BFrag f;
  const unsigned short* q = base + lane * 16;
  f.u[0] = *(const u32x4*)q;
  f.u[1] = *(const u32x4*)(q + 8);
  return f.v;
}

__device__ __forceinline__ f32x8 wmma_bf(bf16x16 a, bf16x16 b, f32x8 c) {
  return __builtin_amdgcn_wmma_f32_16x16x32_bf16(false, a, false, b, (short)0, c, false, false);
}

__device__ __forceinline__ float blockSum256(float v, float* red) {
  int t = threadIdx.x;
  red[t] = v; __syncthreads();
  for (int s = 128; s > 0; s >>= 1) {
    if (t < s) red[t] += red[t + s];
    __syncthreads();
  }
  float r = red[0];
  __syncthreads();
  return r;
}

// ---------------- small prep kernels -----------------------------------------

// per-(b,c) top-10 over L=40  ->  sel_text (B,NQ,HC)
__global__ __launch_bounds__(256) void k_seltext(const float* __restrict__ text,
                                                 float* __restrict__ sel) {
  int t = blockIdx.x * 256 + threadIdx.x;
  if (t >= NB * HC) return;
  int b = t >> 8, c = t & 255;
  float v[NL];
  for (int l = 0; l < NL; ++l) v[l] = text[(size_t)(b * NL + l) * HC + c];
  for (int s = 0; s < NQ; ++s) {
    int bi = 0; float bv = v[0];
    for (int l = 1; l < NL; ++l) if (v[l] > bv) { bv = v[l]; bi = l; }
    sel[(size_t)(b * NQ + s) * HC + c] = bv;
    v[bi] = -3.4e38f;
  }
}

// sine positional embedding, stored TRANSPOSED: kpos[c][n]
__global__ __launch_bounds__(256) void k_posembed(float* __restrict__ kpos) {
  int t = blockIdx.x * 256 + threadIdx.x;
  int n = t >> 8, c = t & 255;
  int i = n >> 7, j = n & 127;
  const float twopi = 6.283185307179586f;
  float yv = (float)(i + 1) * (twopi / (128.0f + 1e-6f));
  float xv = (float)(j + 1) * (twopi / (128.0f + 1e-6f));
  int cc = c & 127;
  float base = (c < 128) ? yv : xv;
  float tp = powf(10000.0f, (float)(cc >> 1) * (1.0f / 64.0f));
  float v = base / tp;
  kpos[((size_t)c << 14) + n] = (cc & 1) ? cosf(v) : sinf(v);
}

// self-attn with value==0 collapses to out_w @ bv + out_b;  x1 = LN1(that)
__global__ __launch_bounds__(256) void k_x1(const float* __restrict__ sa_in_b,
                                            const float* __restrict__ sa_out_w,
                                            const float* __restrict__ sa_out_b,
                                            const float* __restrict__ g1,
                                            const float* __restrict__ b1,
                                            float* __restrict__ x1) {
  __shared__ float red[256];
  int c = threadIdx.x;
  float acc = sa_out_b[c];
  const float* w = sa_out_w + (size_t)c * HC;
  for (int j = 0; j < HC; ++j) acc += w[j] * sa_in_b[2 * HC + j];
  float mu = blockSum256(acc, red) * (1.0f / HC);
  float d = acc - mu;
  float var = blockSum256(d * d, red) * (1.0f / HC);
  x1[c] = d * rsqrtf(var + LNEPS) * g1[c] + b1[c];
}

// Q = (x1 + sel_text) @ wq^T + bq
__global__ __launch_bounds__(256) void k_qproj(const float* __restrict__ sel,
                                               const float* __restrict__ x1,
                                               const float* __restrict__ ca_in_w,
                                               const float* __restrict__ ca_in_b,
                                               float* __restrict__ Qb) {
  __shared__ float qin[HC];
  int bq = blockIdx.x, c = threadIdx.x;
  qin[c] = x1[c] + sel[(size_t)bq * HC + c];
  __syncthreads();
  const float* w = ca_in_w + (size_t)c * HC;   // wq row c
  float acc = ca_in_b[c];
  for (int j = 0; j < HC; ++j) acc += qin[j] * w[j];
  Qb[(size_t)bq * HC + c] = acc;
}

// fold wk into Q: Qk[b][r][c] = sum_d Q[b,q,h*32+d]*wk[h*32+d][c]; pack WMMA A-frags
__global__ __launch_bounds__(256) void k_qkfold(const float* __restrict__ Qb,
                                                const float* __restrict__ ca_in_w,
                                                const float* __restrict__ ca_in_b,
                                                unsigned short* __restrict__ qkfrag,
                                                float* __restrict__ Qbk) {
  __shared__ float Qh[32];
  int br = blockIdx.x;
  int b = br / R80, r = br % R80;
  int h = r / NQ, q = r % NQ;
  int tid = threadIdx.x;
  if (tid < 32) Qh[tid] = Qb[(size_t)(b * NQ + q) * HC + h * 32 + tid];
  __syncthreads();
  int c = tid;
  const float* wk = ca_in_w + HC * HC;   // rows 256..511
  float acc = 0.0f;
  for (int d = 0; d < 32; ++d) acc += Qh[d] * wk[(size_t)(h * 32 + d) * HC + c];
  // fragment packing:  (r,c) -> (tile t, kc, lane, elem)
  int t = r >> 4, lr = r & 15;
  int kc = c >> 5, off = c & 31;
  int hi = off >> 4, o16 = off & 15;
  int hf = o16 >> 3, e = hi * 8 + (o16 & 7);
  int lane = hf * 16 + lr;
  qkfrag[(size_t)b * QKFRAG_B + (size_t)((t * 8 + kc) * 32 + lane) * 16 + e] = f2bf(acc);
  if (tid == 0) {
    float s = 0.0f;
    for (int d = 0; d < 32; ++d) s += Qh[d] * ca_in_b[HC + h * 32 + d];
    Qbk[b * R80 + r] = s;
  }
}

// ---------------- heavy WMMA passes ------------------------------------------

// logits for one 128-position chunk (shared between pass1/pass2)
__device__ __forceinline__ void logits_chunk(const unsigned short* Xbf,
                                             const unsigned short* qkbf,
                                             float* Lsm, int wave, int lane) {
  int half = lane >> 4, lm = lane & 15;
  int N0 = wave * 16;
  for (int rt = 0; rt < 5; ++rt) {
    bf16x16 af[8], bf[8];
#pragma unroll
    for (int kc = 0; kc < 8; ++kc)
      af[kc] = ldsFragC(qkbf + (size_t)((rt * 8 + kc) * 32) * 16, lane);
#pragma unroll
    for (int kc = 0; kc < 8; ++kc)
      bf[kc] = ldsFrag(Xbf, 264, N0, kc * 32, lane);
    CFrag acc;
#pragma unroll
    for (int j = 0; j < 8; ++j) acc.f[j] = 0.0f;
#pragma unroll
    for (int kc = 0; kc < 8; ++kc)
      acc.v = wmma_bf(af[kc], bf[kc], acc.v);
#pragma unroll
    for (int j = 0; j < 8; ++j)
      Lsm[(rt * 16 + j + 8 * half) * CK + N0 + lm] = acc.f[j];
  }
}

// pass1: logits for one 128-position chunk -> per-chunk (max, sumexp) per row
__global__ __launch_bounds__(256) void k_pass1(const float* __restrict__ img,
                                               const float* __restrict__ kpos,
                                               const unsigned short* __restrict__ qkfrag,
                                               const float* __restrict__ Qbk,
                                               float* __restrict__ pstat) {
  extern __shared__ char smem[];
  unsigned short* Xbf  = (unsigned short*)smem;               // [128][264] bf16 (padded)
  float*          Lsm  = (float*)(smem + 128 * 264 * 2);      // [80][128]
  unsigned short* qkbf = (unsigned short*)(Lsm + R80 * CK);   // [20480]
  int b = blockIdx.y, ch = blockIdx.x, n0 = ch * CK, tid = threadIdx.x;
  for (int i = tid; i < CK * HC; i += 256) {
    int kk = i & (CK - 1), c = i >> 7;
    float g = img[(((size_t)(b * HC + c)) << 14) + n0 + kk];
    float x = g + kpos[((size_t)c << 14) + n0 + kk];
    Xbf[kk * 264 + c] = f2bf(x);
  }
  {
    const u32x4* src = (const u32x4*)(qkfrag + (size_t)b * QKFRAG_B);
    u32x4* dst = (u32x4*)qkbf;
    for (int i = tid; i < QKFRAG_B / 8; i += 256) dst[i] = src[i];
  }
  __syncthreads();
  int wave = tid >> 5, lane = tid & 31;
  logits_chunk(Xbf, qkbf, Lsm, wave, lane);
  __syncthreads();
  for (int r = tid; r < R80; r += 256) {
    float qb = Qbk[b * R80 + r];
    const float* row = Lsm + r * CK;
    float mx = -3.4e38f;
    for (int k = 0; k < CK; ++k) mx = fmaxf(mx, (row[k] + qb) * SCALE);
    float sm = 0.0f;
    for (int k = 0; k < CK; ++k) sm += expf((row[k] + qb) * SCALE - mx);
    float* ps = pstat + (((size_t)(b * NCH + ch)) * R80 + r) * 2;
    ps[0] = mx; ps[1] = sm;
  }
}

__global__ __launch_bounds__(256) void k_statreduce(const float* __restrict__ pstat,
                                                    float* __restrict__ gstat) {
  int t = blockIdx.x * 256 + threadIdx.x;
  if (t >= NB * R80) return;
  int b = t / R80, r = t % R80;
  float gm = -3.4e38f;
  for (int ch = 0; ch < NCH; ++ch)
    gm = fmaxf(gm, pstat[(((size_t)(b * NCH + ch)) * R80 + r) * 2]);
  float gs = 0.0f;
  for (int ch = 0; ch < NCH; ++ch) {
    const float* ps = pstat + (((size_t)(b * NCH + ch)) * R80 + r) * 2;
    gs += ps[1] * expf(ps[0] - gm);
  }
  gstat[t * 2] = gm; gstat[t * 2 + 1] = gs;
}

// pass2: recompute logits, emit attn-mean to d_out, then attn @ mem via WMMA
__global__ __launch_bounds__(256) void k_pass2(const float* __restrict__ img,
                                               const float* __restrict__ kpos,
                                               const unsigned short* __restrict__ qkfrag,
                                               const float* __restrict__ Qbk,
                                               const float* __restrict__ gstat,
                                               float* __restrict__ Maccum,
                                               float* __restrict__ dout) {
  extern __shared__ char smem[];
  unsigned short* Xbf    = (unsigned short*)smem;                    // [128][264]
  unsigned short* memT   = Xbf + 128 * 264;                          // [256][136]
  float*          Lsm    = (float*)(memT + 256 * 136);               // [80][128]
  unsigned short* attnbf = (unsigned short*)(Lsm + R80 * CK);        // [80][136]
  unsigned short* qkbf   = attnbf + R80 * 136;                       // [20480]
  int b = blockIdx.y, ch = blockIdx.x, n0 = ch * CK, tid = threadIdx.x;
  for (int i = tid; i < CK * HC; i += 256) {
    int kk = i & (CK - 1), c = i >> 7;
    float g = img[(((size_t)(b * HC + c)) << 14) + n0 + kk];
    float x = g + kpos[((size_t)c << 14) + n0 + kk];
    Xbf[kk * 264 + c] = f2bf(x);
    memT[c * 136 + kk] = f2bf(g);
  }
  {
    const u32x4* src = (const u32x4*)(qkfrag + (size_t)b * QKFRAG_B);
    u32x4* dst = (u32x4*)qkbf;
    for (int i = tid; i < QKFRAG_B / 8; i += 256) dst[i] = src[i];
  }
  __syncthreads();
  int wave = tid >> 5, lane = tid & 31, half = lane >> 4, lm = lane & 15;
  logits_chunk(Xbf, qkbf, Lsm, wave, lane);
  __syncthreads();
  for (int i = tid; i < R80 * CK; i += 256) {
    int r = i >> 7, k = i & 127;
    float qb = Qbk[b * R80 + r];
    float gm = gstat[(b * R80 + r) * 2], gs = gstat[(b * R80 + r) * 2 + 1];
    float a = expf((Lsm[r * CK + k] + qb) * SCALE - gm) / gs;
    attnbf[r * 136 + k] = f2bf(a);
  }
  for (int i = tid; i < NQ * CK; i += 256) {
    int q = i >> 7, k = i & 127;
    float s = 0.0f;
    for (int h = 0; h < NHD; ++h) {
      int r = h * NQ + q;
      float qb = Qbk[b * R80 + r];
      float gm = gstat[(b * R80 + r) * 2], gs = gstat[(b * R80 + r) * 2 + 1];
      s += expf((Lsm[r * CK + k] + qb) * SCALE - gm) / gs;
    }
    dout[O4 + (((size_t)(b * NQ + q)) << 14) + n0 + k] = s * 0.125f;
  }
  __syncthreads();
  for (int rt = 0; rt < 5; ++rt)
    for (int g2 = 0; g2 < 2; ++g2) {
      int ct = wave + g2 * 8;
      bf16x16 af[4], bf[4];
#pragma unroll
      for (int kc = 0; kc < 4; ++kc)
        af[kc] = ldsFrag(attnbf, 136, rt * 16, kc * 32, lane);
#pragma unroll
      for (int kc = 0; kc < 4; ++kc)
        bf[kc] = ldsFrag(memT, 136, ct * 16, kc * 32, lane);
      CFrag acc;
#pragma unroll
      for (int j = 0; j < 8; ++j) acc.f[j] = 0.0f;
#pragma unroll
      for (int kc = 0; kc < 4; ++kc)
        acc.v = wmma_bf(af[kc], bf[kc], acc.v);
#pragma unroll
      for (int j = 0; j < 8; ++j)
        atomicAdd(&Maccum[((size_t)(b * R80 + rt * 16 + j + 8 * half)) * HC + ct * 16 + lm],
                  acc.f[j]);
    }
}

// ---------------- tail kernels ------------------------------------------------

__global__ __launch_bounds__(256) void k_gmax(float* __restrict__ dout) {
  int t = blockIdx.x * 256 + threadIdx.x;
  int b = t >> 14, n = t & (HWN - 1);
  float m = -3.4e38f;
  for (int q = 0; q < NQ; ++q)
    m = fmaxf(m, dout[O4 + (((size_t)(b * NQ + q)) << 14) + n]);
  dout[O3 + ((size_t)b << 14) + n] = m;
}

// greedy score+distance point selector (FPS-like), first-index tie-breaking
__global__ __launch_bounds__(256) void k_selector(float* __restrict__ dout,
                                                  int* __restrict__ posinds) {
  extern __shared__ char smem[];
  float* sc = (float*)smem;          // [16384]
  float* md = sc + HWN;              // [16384]
  __shared__ float rv[256];
  __shared__ int ri[256];
  __shared__ int selidx[NQ];
  int b = blockIdx.x, tid = threadIdx.x;
  const float* g = dout + O3 + ((size_t)b << 14);
  for (int i = tid; i < HWN; i += 256) sc[i] = 1.0f / (1.0f + expf(-g[i]));
  __syncthreads();
  const float kD = 0.1f / sqrtf(32768.0f);
  for (int s = 0; s < NQ; ++s) {
    float bv = -3.4e38f; int bi = 1 << 30;
    for (int i = tid; i < HWN; i += 256) {
      float v = (s == 0) ? sc[i] : sc[i] + kD * md[i];
      if (v > bv) { bv = v; bi = i; }
    }
    rv[tid] = bv; ri[tid] = bi;
    __syncthreads();
    for (int st = 128; st > 0; st >>= 1) {
      if (tid < st) {
        float v2 = rv[tid + st]; int i2 = ri[tid + st];
        if (v2 > rv[tid] || (v2 == rv[tid] && i2 < ri[tid])) { rv[tid] = v2; ri[tid] = i2; }
      }
      __syncthreads();
    }
    int idx = ri[0];
    if (tid == 0) selidx[s] = idx;
    float pr = (float)(idx >> 7), pc = (float)(idx & 127);
    for (int i = tid; i < HWN; i += 256) {
      float dr = (float)(i >> 7) - pr, dc = (float)(i & 127) - pc;
      float d = sqrtf(dr * dr + dc * dc);
      md[i] = (s == 0) ? d : fminf(md[i], d);
    }
    if (tid == 0) sc[idx] = -3.4e38f;
    __syncthreads();
  }
  if (tid < NQ) {
    int idx = selidx[tid];
    posinds[b * NQ + tid] = idx;
    dout[O2 + (size_t)(b * NQ + tid) * 2 + 0] = ((float)(idx & 127) + 0.5f) * (1.0f / 128.0f);
    dout[O2 + (size_t)(b * NQ + tid) * 2 + 1] = ((float)(idx >> 7) + 0.5f) * (1.0f / 128.0f);
  }
}

// per-(b,q): wv-fold + out-proj + LN2 + FFN + LN3 + postLN + gather + 3-layer MLP
__global__ __launch_bounds__(256) void k_head(
    const float* __restrict__ Maccum, const float* __restrict__ img,
    const float* __restrict__ x1, const int* __restrict__ posinds,
    const float* __restrict__ ca_in_w, const float* __restrict__ ca_in_b,
    const float* __restrict__ ca_out_w, const float* __restrict__ ca_out_b,
    const float* __restrict__ ln2_g, const float* __restrict__ ln2_b,
    const float* __restrict__ ln3_g, const float* __restrict__ ln3_b,
    const float* __restrict__ post_g, const float* __restrict__ post_b,
    const float* __restrict__ ffn_w1, const float* __restrict__ ffn_b1,
    const float* __restrict__ ffn_w2, const float* __restrict__ ffn_b2,
    const float* __restrict__ mlp_w1, const float* __restrict__ mlp_b1,
    const float* __restrict__ mlp_w2, const float* __restrict__ mlp_b2,
    const float* __restrict__ mlp_w3, const float* __restrict__ mlp_b3,
    float* __restrict__ dout) {
  __shared__ float sh_o[HC];
  __shared__ float sh_x2[HC];
  __shared__ float sh_hid[FFD];
  __shared__ float sh_z[2 * HC];
  __shared__ float sh_h1[HC];
  __shared__ float sh_h2[HC];
  __shared__ float red[256];
  int b = blockIdx.x / NQ, q = blockIdx.x % NQ, c = threadIdx.x;
  int h = c >> 5, r = h * NQ + q;
  const float* Mrow = Maccum + (size_t)(b * R80 + r) * HC;
  const float* wv = ca_in_w + 2 * HC * HC + (size_t)c * HC;
  float acc = ca_in_b[2 * HC + c];
  for (int j = 0; j < HC; ++j) acc += Mrow[j] * wv[j];
  sh_o[c] = acc;
  __syncthreads();
  const float* ow = ca_out_w + (size_t)c * HC;
  float ca = ca_out_b[c];
  for (int j = 0; j < HC; ++j) ca += sh_o[j] * ow[j];
  float y = x1[c] + ca;
  float mu = blockSum256(y, red) * (1.0f / HC);
  float d = y - mu;
  float var = blockSum256(d * d, red) * (1.0f / HC);
  float x2v = d * rsqrtf(var + LNEPS) * ln2_g[c] + ln2_b[c];
  sh_x2[c] = x2v;
  __syncthreads();
  for (int f = c; f < FFD; f += 256) {
    const float* w = ffn_w1 + (size_t)f * HC;
    float a = ffn_b1[f];
    for (int j = 0; j < HC; ++j) a += sh_x2[j] * w[j];
    sh_hid[f] = fmaxf(a, 0.0f);
  }
  __syncthreads();
  const float* w2 = ffn_w2 + (size_t)c * FFD;
  float fo = ffn_b2[c];
  for (int j = 0; j < FFD; ++j) fo += sh_hid[j] * w2[j];
  float y3 = x2v + fo;
  mu = blockSum256(y3, red) * (1.0f / HC);
  d = y3 - mu;
  var = blockSum256(d * d, red) * (1.0f / HC);
  float t3 = d * rsqrtf(var + LNEPS) * ln3_g[c] + ln3_b[c];
  mu = blockSum256(t3, red) * (1.0f / HC);
  d = t3 - mu;
  var = blockSum256(d * d, red) * (1.0f / HC);
  float to = d * rsqrtf(var + LNEPS) * post_g[c] + post_b[c];
  sh_z[c] = to;
  int idx = posinds[b * NQ + q];
  sh_z[HC + c] = img[(((size_t)(b * HC + c)) << 14) + idx];
  __syncthreads();
  const float* m1 = mlp_w1 + (size_t)c * (2 * HC);
  float a1 = mlp_b1[c];
  for (int j = 0; j < 2 * HC; ++j) a1 += sh_z[j] * m1[j];
  sh_h1[c] = fmaxf(a1, 0.0f);
  __syncthreads();
  const float* m2 = mlp_w2 + (size_t)c * HC;
  float a2 = mlp_b2[c];
  for (int j = 0; j < HC; ++j) a2 += sh_h1[j] * m2[j];
  sh_h2[c] = fmaxf(a2, 0.0f);
  __syncthreads();
  const float* m3 = mlp_w3 + (size_t)c * HC;
  float a3 = mlp_b3[c];
  for (int j = 0; j < HC; ++j) a3 += sh_h2[j] * m3[j];
  dout[O1 + (size_t)(b * NQ + q) * HC + c] = a3;
}

// ---------------- launcher ----------------------------------------------------

extern "C" void kernel_launch(void* const* d_in, const int* in_sizes, int n_in,
                              void* d_out, int out_size, void* d_ws, size_t ws_size,
                              hipStream_t stream) {
  (void)in_sizes; (void)n_in; (void)out_size; (void)ws_size;
  const float* text     = (const float*)d_in[0];
  const float* img      = (const float*)d_in[2];
  const float* sa_in_b  = (const float*)d_in[4];
  const float* sa_out_w = (const float*)d_in[5];
  const float* sa_out_b = (const float*)d_in[6];
  const float* ca_in_w  = (const float*)d_in[7];
  const float* ca_in_b  = (const float*)d_in[8];
  const float* ca_out_w = (const float*)d_in[9];
  const float* ca_out_b = (const float*)d_in[10];
  const float* ln1_g = (const float*)d_in[11];
  const float* ln1_b = (const float*)d_in[12];
  const float* ln2_g = (const float*)d_in[13];
  const float* ln2_b = (const float*)d_in[14];
  const float* ln3_g = (const float*)d_in[15];
  const float* ln3_b = (const float*)d_in[16];
  const float* post_g = (const float*)d_in[17];
  const float* post_b = (const float*)d_in[18];
  const float* ffn_w1 = (const float*)d_in[19];
  const float* ffn_b1 = (const float*)d_in[20];
  const float* ffn_w2 = (const float*)d_in[21];
  const float* ffn_b2 = (const float*)d_in[22];
  const float* mlp_w1 = (const float*)d_in[23];
  const float* mlp_b1 = (const float*)d_in[24];
  const float* mlp_w2 = (const float*)d_in[25];
  const float* mlp_b2 = (const float*)d_in[26];
  const float* mlp_w3 = (const float*)d_in[27];
  const float* mlp_b3 = (const float*)d_in[28];
  float* dout = (float*)d_out;

  // workspace partition (floats); ~24 MB total
  float* ws       = (float*)d_ws;
  float* key_pos  = ws;                         // 4,194,304  [c][n]
  float* sel_text = key_pos + 4194304;          // 81,920
  float* x1       = sel_text + 81920;           // 256
  float* Qb       = x1 + 256;                   // 81,920
  float* Qbk      = Qb + 81920;                 // 2,560
  float* pstat    = Qbk + 2560;                 // 655,360
  float* gstat    = pstat + 655360;             // 5,120
  float* Maccum   = gstat + 5120;               // 655,360
  int*   posinds  = (int*)(Maccum + 655360);    // 320 ints
  unsigned short* qkfrag = (unsigned short*)(posinds + 320);  // 655,360 u16

  k_seltext<<<(NB * HC) / 256, 256, 0, stream>>>(text, sel_text);
  k_posembed<<<(HWN * HC) / 256, 256, 0, stream>>>(key_pos);
  k_x1<<<1, 256, 0, stream>>>(sa_in_b, sa_out_w, sa_out_b, ln1_g, ln1_b, x1);
  k_qproj<<<NB * NQ, 256, 0, stream>>>(sel_text, x1, ca_in_w, ca_in_b, Qb);
  k_qkfold<<<NB * R80, 256, 0, stream>>>(Qb, ca_in_w, ca_in_b, qkfrag, Qbk);

  hipMemsetAsync((void*)Maccum, 0, (size_t)NB * R80 * HC * sizeof(float), stream);

  dim3 gAttn(NCH, NB);
  size_t lds1 = (size_t)128 * 264 * 2 + (size_t)R80 * CK * 4 +
                (size_t)QKFRAG_B * 2;                                               // 149,504
  size_t lds2 = (size_t)128 * 264 * 2 + (size_t)256 * 136 * 2 +
                (size_t)R80 * CK * 4 + (size_t)R80 * 136 * 2 +
                (size_t)QKFRAG_B * 2;                                               // 240,896
  k_pass1<<<gAttn, 256, lds1, stream>>>(img, key_pos, qkfrag, Qbk, pstat);
  k_statreduce<<<(NB * R80 + 255) / 256, 256, 0, stream>>>(pstat, gstat);
  k_pass2<<<gAttn, 256, lds2, stream>>>(img, key_pos, qkfrag, Qbk, gstat, Maccum, dout);

  k_gmax<<<(NB * HWN) / 256, 256, 0, stream>>>(dout);
  k_selector<<<NB, 256, (size_t)2 * HWN * 4, stream>>>(dout, posinds);
  k_head<<<NB * NQ, 256, 0, stream>>>(Maccum, img, x1, posinds,
                                      ca_in_w, ca_in_b, ca_out_w, ca_out_b,
                                      ln2_g, ln2_b, ln3_g, ln3_b, post_g, post_b,
                                      ffn_w1, ffn_b1, ffn_w2, ffn_b2,
                                      mlp_w1, mlp_b1, mlp_w2, mlp_b2, mlp_w3, mlp_b3,
                                      dout);
}
